// NeuralCDE_60035052863494
// MI455X (gfx1250) — matrix operand order repaired and checked by reading, not compiled, not run on hardware
//
#include <hip/hip_runtime.h>

typedef __attribute__((ext_vector_type(16))) _Float16 v16h;
typedef __attribute__((ext_vector_type(8)))  _Float16 v8h;
typedef __attribute__((ext_vector_type(8)))  float    v8f;

#define Bsz 256
#define Tsz 1024
#define Csz 8
#define Hsz 64
#define Wsz 128

// One block = one 16-row batch tile, 8 waves (wave32), persistent over all 1023 steps.
// Transposed math: H1^T(128x16) = W1^T(128x64) @ z^T(64x16)
//                  O^T (512x16) = W2^T(512x128) @ H1^T(128x16)
// Batch is the WMMA N dim; the einsum reduction over c is purely in-lane.
// Loop-invariant weight A-fragments live in VGPRs (144 regs of f16), gathered
// once from global; only the dynamic z^T / H1^T staging goes through LDS.
__global__ __launch_bounds__(256) void cde_scan_kernel(
    const float* __restrict__ Ca, const float* __restrict__ Cb,
    const float* __restrict__ Cc, const float* __restrict__ Cd,
    const float* __restrict__ Winit, const float* __restrict__ binit,
    const float* __restrict__ W1g, const float* __restrict__ b1g,
    const float* __restrict__ W2g, const float* __restrict__ b2g,
    float* __restrict__ out)
{
  __shared__ float b1s[128];
  __shared__ float b2s[512];
  __shared__ __align__(32) _Float16 hbuf[16 * 128];  // H1^T staging [n][128] (B-frag layout)
  __shared__ __align__(32) _Float16 zbuf[16 * 64];   // z^T  staging [n][64]  (B-frag layout)

  const int tid  = threadIdx.x;
  const int lane = tid & 31;
  const int w    = tid >> 5;      // wave id 0..7
  const int n    = lane & 15;     // batch column within tile (WMMA N)
  const int hf   = lane >> 4;     // lane half (0/1)
  const int bBase = blockIdx.x * 16;
  const int b     = bBase + n;

  // ---- biases into LDS (read every step, off the critical path) ----
  for (int i = tid; i < 128; i += 256) b1s[i] = b1g[i];
  for (int i = tid; i < 512; i += 256) b2s[i] = b2g[i];

  // ---- gather loop-invariant weight A-fragments straight into VGPRs ----
  // A-frag element e of lane (hf,n): K = kc*32 + 16*hf + e
  v16h w1f[2];
  {
    int m = 16 * w + n;                 // row of W1^T (hidden unit j), mt = w
    #pragma unroll
    for (int kc = 0; kc < 2; ++kc) {
      v16h f;
      #pragma unroll
      for (int e = 0; e < 16; ++e) {
        int K = kc * 32 + 16 * hf + e;     // K < 64
        f[e] = (_Float16)W1g[K * Wsz + m];                    // W1^T[m][K] = W1[K][m]
      }
      w1f[kc] = f;
    }
  }
  v16h w2f[16];
  #pragma unroll
  for (int tt = 0; tt < 4; ++tt) {
    int m = 16 * (4 * w + tt) + n;      // row of W2^T (output j2), tile t2 = 4w+tt
    #pragma unroll
    for (int kc = 0; kc < 4; ++kc) {
      v16h f;
      #pragma unroll
      for (int e = 0; e < 16; ++e) {
        int K = kc * 32 + 16 * hf + e;     // K < 128
        f[e] = (_Float16)W2g[K * (Hsz * Csz) + m];            // W2^T[m][K] = W2[K][m]
      }
      w2f[tt * 4 + kc] = f;
    }
  }

  // ---- z0 = x0 @ W_init + b_init ----
  // wave w owns z rows hh in [8w, 8w+8): lane holds hh = 8w + 2*i + hf, i=0..3
  float zst[4];
  {
    float a0[8];
    #pragma unroll
    for (int c = 0; c < 8; ++c) a0[c] = Ca[(long)b * (Tsz - 1) * Csz + c];
    #pragma unroll
    for (int i = 0; i < 4; ++i) {
      int hh = 8 * w + 2 * i + hf;
      float z = binit[hh];
      #pragma unroll
      for (int c = 0; c < 8; ++c) z += a0[c] * Winit[c * Hsz + hh];
      zst[i] = z;
      zbuf[n * 64 + hh] = (_Float16)z;
      out[(long)b * (Tsz * Hsz) + hh] = z;
    }
  }

  // ---- dx prefetch for step 0 (consumed one full step later) ----
  float4 pb0, pb1, pc0, pc1, pd0, pd1;
  {
    long cbase = (long)b * (Tsz - 1) * Csz;
    pb0 = *(const float4*)(Cb + cbase); pb1 = *(const float4*)(Cb + cbase + 4);
    pc0 = *(const float4*)(Cc + cbase); pc1 = *(const float4*)(Cc + cbase + 4);
    pd0 = *(const float4*)(Cd + cbase); pd1 = *(const float4*)(Cd + cbase + 4);
  }

  __syncthreads();

  // ---------------- the scan ----------------
  for (int t = 0; t < Tsz - 1; ++t) {
    // ---- GEMM1: H1^T tile mt = w, A operands in registers ----
    v8f acc;
    {
      float4 u0 = *(const float4*)(b1s + 16 * w + 8 * hf);
      float4 u1 = *(const float4*)(b1s + 16 * w + 8 * hf + 4);
      acc[0] = u0.x; acc[1] = u0.y; acc[2] = u0.z; acc[3] = u0.w;
      acc[4] = u1.x; acc[5] = u1.y; acc[6] = u1.z; acc[7] = u1.w;
    }
    #pragma unroll
    for (int kc = 0; kc < 2; ++kc) {
      v16h bz = *(const v16h*)(zbuf + n * 64 + kc * 32 + 16 * hf);
      acc = __builtin_amdgcn_wmma_f32_16x16x32_f16(false, w1f[kc], false, bz,
                                                   (short)0, acc, false, false);
    }
    // relu -> f16 -> hbuf (contiguous 8 halfs, one ds_store_b128)
    {
      v8h hv;
      #pragma unroll
      for (int r = 0; r < 8; ++r) {
        float xr = acc[r];
        hv[r] = (_Float16)(xr > 0.0f ? xr : 0.0f);
      }
      *(v8h*)(hbuf + n * 128 + 16 * w + 8 * hf) = hv;
    }
    __syncthreads();

    // ---- H1^T B-fragments (the only cross-wave data) ----
    v16h bh[4];
    #pragma unroll
    for (int kc = 0; kc < 4; ++kc)
      bh[kc] = *(const v16h*)(hbuf + n * 128 + kc * 32 + 16 * hf);

    // ---- consume prefetched dx, immediately prefetch next step ----
    float dxv[8];
    dxv[0] = pb0.x + pc0.x + 0.75f * pd0.x;  dxv[1] = pb0.y + pc0.y + 0.75f * pd0.y;
    dxv[2] = pb0.z + pc0.z + 0.75f * pd0.z;  dxv[3] = pb0.w + pc0.w + 0.75f * pd0.w;
    dxv[4] = pb1.x + pc1.x + 0.75f * pd1.x;  dxv[5] = pb1.y + pc1.y + 0.75f * pd1.y;
    dxv[6] = pb1.z + pc1.z + 0.75f * pd1.z;  dxv[7] = pb1.w + pc1.w + 0.75f * pd1.w;
    {
      int tn = (t < Tsz - 3) ? (t + 1) : (Tsz - 3);      // clamp, avoid OOB on last step
      long nb = ((long)b * (Tsz - 1) + tn) * Csz;
      pb0 = *(const float4*)(Cb + nb); pb1 = *(const float4*)(Cb + nb + 4);
      pc0 = *(const float4*)(Cc + nb); pc1 = *(const float4*)(Cc + nb + 4);
      pd0 = *(const float4*)(Cd + nb); pd1 = *(const float4*)(Cd + nb + 4);
    }

    // ---- GEMM2: O^T tiles t2 = 4w+tt; tanh; in-lane einsum; z update ----
    #pragma unroll
    for (int tt = 0; tt < 4; ++tt) {
      int t2 = 4 * w + tt;
      v8f acc2;
      {
        float4 u0 = *(const float4*)(b2s + 16 * t2 + 8 * hf);
        float4 u1 = *(const float4*)(b2s + 16 * t2 + 8 * hf + 4);
        acc2[0] = u0.x; acc2[1] = u0.y; acc2[2] = u0.z; acc2[3] = u0.w;
        acc2[4] = u1.x; acc2[5] = u1.y; acc2[6] = u1.z; acc2[7] = u1.w;
      }
      #pragma unroll
      for (int kc = 0; kc < 4; ++kc) {
        acc2 = __builtin_amdgcn_wmma_f32_16x16x32_f16(false, w2f[tt * 4 + kc], false,
                                                      bh[kc], (short)0, acc2, false, false);
      }
      // lane holds o[j][n], j = 16*t2 + 8*hf + r -> hh = 8w+2tt+hf, c = r (in-lane!)
      float delta = 0.0f;
      #pragma unroll
      for (int r = 0; r < 8; ++r) {
        float x  = acc2[r];
        float e2 = __expf(2.0f * x);
        float th = 1.0f - 2.0f / (e2 + 1.0f);   // tanh(x)
        delta += th * dxv[r];
      }
      zst[tt] += delta;
      int hh = 8 * w + 2 * tt + hf;
      zbuf[n * 64 + hh] = (_Float16)zst[tt];
      out[(long)b * (Tsz * Hsz) + (long)(t + 1) * Hsz + hh] = zst[tt];
    }
    __syncthreads();   // zbuf/hbuf consistent before next step
  }
}

extern "C" void kernel_launch(void* const* d_in, const int* in_sizes, int n_in,
                              void* d_out, int out_size, void* d_ws, size_t ws_size,
                              hipStream_t stream) {
  (void)in_sizes; (void)n_in; (void)d_ws; (void)ws_size; (void)out_size;
  const float* Ca    = (const float*)d_in[0];
  const float* Cb    = (const float*)d_in[1];
  const float* Cc    = (const float*)d_in[2];
  const float* Cd    = (const float*)d_in[3];
  const float* Winit = (const float*)d_in[4];
  const float* binit = (const float*)d_in[5];
  const float* W1    = (const float*)d_in[6];
  const float* b1    = (const float*)d_in[7];
  const float* W2    = (const float*)d_in[8];
  const float* b2    = (const float*)d_in[9];
  float* out = (float*)d_out;

  cde_scan_kernel<<<dim3(Bsz / 16), dim3(256), 0, stream>>>(
      Ca, Cb, Cc, Cd, Winit, binit, W1, b1, W2, b2, out);
}
